// Custom_10539849744520
// MI455X (gfx1250) — matrix operand order repaired and checked by reading, not compiled, not run on hardware
//
#include <hip/hip_runtime.h>
#include <math.h>

// ---------------- types for WMMA ----------------
typedef __attribute__((ext_vector_type(16))) __bf16          v16bf;
typedef __attribute__((ext_vector_type(16))) unsigned short  v16us;
typedef __attribute__((ext_vector_type(8)))  float           v8f;

union U16 { v16us u; v16bf b; };

#define B_    8
#define S_    32
#define D_    512
#define H_    8
#define HD_   64
#define L_    2
#define G4_   2048     // 4*D
#define QKV_  1536     // 3*D
#define SB_   256      // S_*B_
#define NEGINF_ (-1000000000.0f)

__device__ __forceinline__ unsigned short f2bf(float f) {
  unsigned int u = __float_as_uint(f);
  unsigned int r = (u + 0x7FFFu + ((u >> 16) & 1u)) >> 16;  // round-to-nearest-even
  return (unsigned short)r;
}
__device__ __forceinline__ float sigmf(float x) { return 1.0f / (1.0f + expf(-x)); }

// ---------------- weight conversion f32 -> bf16 ----------------
__global__ void to_bf16(const float* __restrict__ src, unsigned short* __restrict__ dst, int n) {
  int i = blockIdx.x * blockDim.x + threadIdx.x;
  if (i < n) dst[i] = f2bf(src[i]);
}

// ---------------- WMMA GEMM: C = epi(A[M,K]f32 @ W[N,K]bf16 ^T) ----------------
// One wave computes a 16x64 tile of C (4 WMMA accumulators).
// A rows beyond Mreal are read from a clamped (in-bounds) row; the resulting
// garbage only reaches C rows that are never stored, so no masking is needed.
// flags: 1 = accumulate into existing C, 2 = ReLU
__global__ __launch_bounds__(32)
void gemm_wmma(const float* __restrict__ A, const unsigned short* __restrict__ W,
               const float* __restrict__ b1, const float* __restrict__ b2,
               const float* __restrict__ pe, const float* __restrict__ resid,
               float* __restrict__ C, int Mreal, int N, int K,
               float alpha, int flags, int peDiv)
{
  const int tilesN = N >> 6;               // N/64
  const int tm     = blockIdx.x / tilesN;
  const int tn     = blockIdx.x % tilesN;
  const int lane   = threadIdx.x & 31;
  const int l15    = lane & 15;
  const int kgrp   = lane >> 4;            // 0 or 1
  const int rowA   = tm * 16 + l15;
  const int rowS   = (rowA < Mreal) ? rowA : (Mreal - 1);   // clamp, no masking

  const float*          Abase = A + (size_t)rowS * K + kgrp * 16;
  const unsigned short* Wbase = W + (size_t)(tn * 64 + l15) * K + kgrp * 16;

  v8f acc[4] = {};

#pragma unroll 2
  for (int k0 = 0; k0 < K; k0 += 32) {
    // A fragment (16x32 bf16): elems 0..15 = K k0+kgrp*16+e
    const float* ap = Abase + k0;
    float4 a0 = *(const float4*)(ap);
    float4 a1 = *(const float4*)(ap + 4);
    float4 a2 = *(const float4*)(ap + 8);
    float4 a3 = *(const float4*)(ap + 12);
    v16bf af;
    af[0]  = (__bf16)a0.x; af[1]  = (__bf16)a0.y; af[2]  = (__bf16)a0.z; af[3]  = (__bf16)a0.w;
    af[4]  = (__bf16)a1.x; af[5]  = (__bf16)a1.y; af[6]  = (__bf16)a1.z; af[7]  = (__bf16)a1.w;
    af[8]  = (__bf16)a2.x; af[9]  = (__bf16)a2.y; af[10] = (__bf16)a2.z; af[11] = (__bf16)a2.w;
    af[12] = (__bf16)a3.x; af[13] = (__bf16)a3.y; af[14] = (__bf16)a3.z; af[15] = (__bf16)a3.w;

#pragma unroll
    for (int t = 0; t < 4; ++t) {
      // B fragment (32x16 bf16): lane holds W row (tn*64 + t*16 + l15), K = k0+kgrp*16+(0..15)
      U16 ub;
      ub.u = *(const v16us*)(Wbase + (size_t)t * 16 * K + k0);
      acc[t] = __builtin_amdgcn_wmma_f32_16x16x32_bf16(
          false, af, false, ub.b, (short)0, acc[t], false, false);
    }
  }

  // epilogue
#pragma unroll
  for (int t = 0; t < 4; ++t) {
    const int col = tn * 64 + t * 16 + l15;
    float bias = 0.0f;
    if (b1) bias += b1[col];
    if (b2) bias += b2[col];
#pragma unroll
    for (int v = 0; v < 8; ++v) {
      const int row = tm * 16 + v + 8 * kgrp;
      if (row < Mreal) {
        float val = (acc[t][v] + bias) * alpha;
        if (pe)        val += pe[(size_t)(row / peDiv) * N + col];
        if (resid)     val += resid[(size_t)row * N + col];
        if (flags & 1) val += C[(size_t)row * N + col];
        if (flags & 2) val = fmaxf(val, 0.0f);
        C[(size_t)row * N + col] = val;
      }
    }
  }
}

// ---------------- pointwise / helper kernels ----------------
__global__ void fill_zero(float* p, int n) {
  int i = blockIdx.x * blockDim.x + threadIdx.x;
  if (i < n) p[i] = 0.0f;
}

__global__ void copy_rows(const float* __restrict__ src, int sstride,
                          float* __restrict__ dst, int dstride, int rows, int cols) {
  int idx = blockIdx.x * blockDim.x + threadIdx.x;
  if (idx >= rows * cols) return;
  int r = idx / cols, c = idx % cols;
  dst[(size_t)r * dstride + c] = src[(size_t)r * sstride + c];
}

__global__ void lstm_point(const float* __restrict__ g, float* __restrict__ h, float* __restrict__ c) {
  int idx = blockIdx.x * blockDim.x + threadIdx.x;      // B_*D_ = 4096
  if (idx >= B_ * D_) return;
  int b = idx >> 9, j = idx & 511;
  const float* gp = g + (size_t)b * G4_;
  float gi = gp[j], gf = gp[512 + j], gg = gp[1024 + j], go = gp[1536 + j];
  float cn = sigmf(gf) * c[idx] + sigmf(gi) * tanhf(gg);
  float hn = sigmf(go) * tanhf(cn);
  c[idx] = cn; h[idx] = hn;
}

__global__ void pe_fill(float* pe) {
  int idx = blockIdx.x * blockDim.x + threadIdx.x;      // S_*D_
  if (idx >= S_ * D_) return;
  int d = idx & 511, pos = idx >> 9;
  int j = d >> 1;
  float div = expf(-(float)(2 * j) * (9.210340371976184f / 512.0f));  // ln(1e4)/D
  float ang = (float)pos * div;
  pe[idx] = (d & 1) ? cosf(ang) : sinf(ang);
}

// scores[b,h,q,k] = (q . k)/8 (+mask)
__global__ void attn_scores(const float* __restrict__ Q, int ldq, int cq,
                            const float* __restrict__ Km, int ldk, int ck,
                            float* __restrict__ Sc, int useMask) {
  int idx = blockIdx.x * blockDim.x + threadIdx.x;      // B*H*S*S = 65536
  if (idx >= B_ * H_ * S_ * S_) return;
  int k = idx & 31, q = (idx >> 5) & 31, h = (idx >> 10) & 7, b = idx >> 13;
  const float* qp = Q  + (size_t)(q * B_ + b) * ldq + cq + h * HD_;
  const float* kp = Km + (size_t)(k * B_ + b) * ldk + ck + h * HD_;
  float s = 0.0f;
#pragma unroll 8
  for (int d = 0; d < HD_; ++d) s += qp[d] * kp[d];
  s *= 0.125f;
  if (useMask && k > q) s += NEGINF_;
  Sc[idx] = s;
}

__global__ __launch_bounds__(32) void softmax32(float* __restrict__ Sc) {
  int row = blockIdx.x, lane = threadIdx.x;
  float v = Sc[(size_t)row * 32 + lane];
  float m = v;
  for (int off = 16; off; off >>= 1) m = fmaxf(m, __shfl_xor(m, off, 32));
  float e = expf(v - m);
  float s = e;
  for (int off = 16; off; off >>= 1) s += __shfl_xor(s, off, 32);
  Sc[(size_t)row * 32 + lane] = e / s;
}

__global__ void attn_apply(const float* __restrict__ Sc,
                           const float* __restrict__ V, int ldv, int cv,
                           float* __restrict__ O) {
  int idx = blockIdx.x * blockDim.x + threadIdx.x;      // B*H*S*HD = 131072
  if (idx >= B_ * H_ * S_ * HD_) return;
  int d = idx & 63, q = (idx >> 6) & 31, h = (idx >> 11) & 7, b = idx >> 14;
  const float* ap = Sc + ((size_t)(b * H_ + h) * S_ + q) * S_;
  float acc = 0.0f;
#pragma unroll 8
  for (int k = 0; k < S_; ++k)
    acc += ap[k] * V[(size_t)(k * B_ + b) * ldv + cv + h * HD_ + d];
  O[(size_t)(q * B_ + b) * D_ + h * HD_ + d] = acc;
}

__global__ __launch_bounds__(32)
void layernorm512(float* __restrict__ X, const float* __restrict__ gw,
                  const float* __restrict__ bw, int rowOff) {
  int row = rowOff + blockIdx.x, lane = threadIdx.x;
  float* xp = X + (size_t)row * D_;
  float vals[16];
  float s = 0.0f;
#pragma unroll
  for (int i = 0; i < 16; ++i) { vals[i] = xp[lane + i * 32]; s += vals[i]; }
  for (int off = 16; off; off >>= 1) s += __shfl_xor(s, off, 32);
  float m = s * (1.0f / 512.0f);
  float vs = 0.0f;
#pragma unroll
  for (int i = 0; i < 16; ++i) { float d = vals[i] - m; vs += d * d; }
  for (int off = 16; off; off >>= 1) vs += __shfl_xor(vs, off, 32);
  float inv = rsqrtf(vs * (1.0f / 512.0f) + 1e-5f);
#pragma unroll
  for (int i = 0; i < 16; ++i) {
    int c = lane + i * 32;
    xp[c] = (vals[i] - m) * inv * gw[c] + bw[c];
  }
}

// dec (t*B+b, D) -> out (b, t, D)
__global__ void out_transpose(const float* __restrict__ dec, float* __restrict__ out) {
  int idx = blockIdx.x * blockDim.x + threadIdx.x;      // B*S*D = 131072
  if (idx >= B_ * S_ * D_) return;
  int j = idx & 511, t = (idx >> 9) & 31, b = idx >> 14;
  out[idx] = dec[((size_t)t * B_ + b) * D_ + j];
}

// ---------------- host orchestration ----------------
static inline void gemm(hipStream_t st, const float* A, const unsigned short* W,
                        const float* b1, const float* b2, const float* pe,
                        const float* resid, float* C, int Mreal, int N, int K,
                        float alpha, int flags, int peDiv) {
  int tilesM = (Mreal + 15) / 16, tilesN = N / 64;
  gemm_wmma<<<dim3(tilesM * tilesN), dim3(32), 0, st>>>(A, W, b1, b2, pe, resid, C,
                                                        Mreal, N, K, alpha, flags, peDiv);
}

extern "C" void kernel_launch(void* const* d_in, const int* in_sizes, int n_in,
                              void* d_out, int out_size, void* d_ws, size_t ws_size,
                              hipStream_t stream) {
  (void)in_sizes; (void)n_in; (void)out_size; (void)ws_size;
  const float* src    = (const float*)d_in[0];
  const float* tgt    = (const float*)d_in[1];
  const float* wih    = (const float*)d_in[2];
  const float* whh    = (const float*)d_in[3];
  const float* bih    = (const float*)d_in[4];
  const float* bhh    = (const float*)d_in[5];
  const float* projw  = (const float*)d_in[6];
  const float* projb  = (const float*)d_in[7];
  const float* encw   = (const float*)d_in[8];
  const float* encb   = (const float*)d_in[9];
  const float* outw   = (const float*)d_in[10];
  const float* outb   = (const float*)d_in[11];
  const float* saw    = (const float*)d_in[12];
  const float* sab    = (const float*)d_in[13];
  const float* saow   = (const float*)d_in[14];
  const float* saob   = (const float*)d_in[15];
  const float* caw    = (const float*)d_in[16];
  const float* cab    = (const float*)d_in[17];
  const float* caow   = (const float*)d_in[18];
  const float* caob   = (const float*)d_in[19];
  const float* w1     = (const float*)d_in[20];
  const float* b1p    = (const float*)d_in[21];
  const float* w2     = (const float*)d_in[22];
  const float* b2p    = (const float*)d_in[23];
  const float* n1g    = (const float*)d_in[24];
  const float* n1b    = (const float*)d_in[25];
  const float* n2g    = (const float*)d_in[26];
  const float* n2b    = (const float*)d_in[27];
  const float* n3g    = (const float*)d_in[28];
  const float* n3b    = (const float*)d_in[29];
  const float* nfg    = (const float*)d_in[30];
  const float* nfb    = (const float*)d_in[31];
  float* outp = (float*)d_out;

  // bump allocator on workspace
  char* base = (char*)d_ws; size_t off = 0;
  auto alloc = [&](size_t bytes) -> void* {
    off = (off + 255) & ~(size_t)255;
    void* p = base + off; off += bytes; return p;
  };

  // bf16 weight copies
  unsigned short* wih_b  = (unsigned short*)alloc((size_t)L_ * G4_ * D_ * 2);
  unsigned short* whh_b  = (unsigned short*)alloc((size_t)L_ * G4_ * D_ * 2);
  unsigned short* proj_b = (unsigned short*)alloc((size_t)D_ * D_ * 2);
  unsigned short* enc_b  = (unsigned short*)alloc((size_t)D_ * D_ * 2);
  unsigned short* out_b  = (unsigned short*)alloc((size_t)D_ * D_ * 2);
  unsigned short* saw_b  = (unsigned short*)alloc((size_t)L_ * QKV_ * D_ * 2);
  unsigned short* saow_b = (unsigned short*)alloc((size_t)L_ * D_ * D_ * 2);
  unsigned short* caw_b  = (unsigned short*)alloc((size_t)L_ * QKV_ * D_ * 2);
  unsigned short* caow_b = (unsigned short*)alloc((size_t)L_ * D_ * D_ * 2);
  unsigned short* w1_b   = (unsigned short*)alloc((size_t)L_ * D_ * D_ * 2);
  unsigned short* w2_b   = (unsigned short*)alloc((size_t)L_ * D_ * D_ * 2);

  // fp32 activation buffers
  float* h     = (float*)alloc((size_t)L_ * B_ * D_ * 4);
  float* c     = (float*)alloc((size_t)L_ * B_ * D_ * 4);
  float* g     = (float*)alloc((size_t)B_ * G4_ * 4);
  float* xstep = (float*)alloc((size_t)B_ * D_ * 4);
  float* mem   = (float*)alloc((size_t)SB_ * D_ * 4);
  float* dec   = (float*)alloc((size_t)SB_ * D_ * 4);
  float* x     = (float*)alloc((size_t)SB_ * D_ * 4);
  float* qkv   = (float*)alloc((size_t)SB_ * QKV_ * 4);
  float* cakv  = (float*)alloc((size_t)L_ * SB_ * 2 * D_ * 4);
  float* sc    = (float*)alloc((size_t)B_ * H_ * S_ * S_ * 4);
  float* attno = (float*)alloc((size_t)SB_ * D_ * 4);
  float* ff    = (float*)alloc((size_t)SB_ * D_ * 4);
  float* pe    = (float*)alloc((size_t)S_ * D_ * 4);
  float* nxt   = (float*)alloc((size_t)B_ * D_ * 4);
  // safety pad so clamped-row A reads past a buffer end stay inside the workspace
  (void)alloc((size_t)16 * D_ * 4);

  auto cvt = [&](const float* s, unsigned short* d, int n) {
    to_bf16<<<dim3((n + 255) / 256), dim3(256), 0, stream>>>(s, d, n);
  };
  cvt(wih,   wih_b,  L_ * G4_ * D_);
  cvt(whh,   whh_b,  L_ * G4_ * D_);
  cvt(projw, proj_b, D_ * D_);
  cvt(encw,  enc_b,  D_ * D_);
  cvt(outw,  out_b,  D_ * D_);
  cvt(saw,   saw_b,  L_ * QKV_ * D_);
  cvt(saow,  saow_b, L_ * D_ * D_);
  cvt(caw,   caw_b,  L_ * QKV_ * D_);
  cvt(caow,  caow_b, L_ * D_ * D_);
  cvt(w1,    w1_b,   L_ * D_ * D_);
  cvt(w2,    w2_b,   L_ * D_ * D_);

  fill_zero<<<dim3((L_ * B_ * D_ + 255) / 256), dim3(256), 0, stream>>>(h, L_ * B_ * D_);
  fill_zero<<<dim3((L_ * B_ * D_ + 255) / 256), dim3(256), 0, stream>>>(c, L_ * B_ * D_);
  fill_zero<<<dim3((SB_ * D_ + 255) / 256), dim3(256), 0, stream>>>(dec, SB_ * D_);
  pe_fill<<<dim3((S_ * D_ + 255) / 256), dim3(256), 0, stream>>>(pe);

  // -------- LSTM encoder over src (32 steps) --------
  for (int t = 0; t < S_; ++t) {
    copy_rows<<<dim3((B_ * D_ + 255) / 256), dim3(256), 0, stream>>>(
        src + (size_t)t * D_, S_ * D_, xstep, D_, B_, D_);
    const float* inp = xstep;
    for (int l = 0; l < L_; ++l) {
      gemm(stream, inp, wih_b + (size_t)l * G4_ * D_, bih + l * G4_, bhh + l * G4_,
           nullptr, nullptr, g, B_, G4_, D_, 1.0f, 0, 1);
      gemm(stream, h + (size_t)l * B_ * D_, whh_b + (size_t)l * G4_ * D_,
           nullptr, nullptr, nullptr, nullptr, g, B_, G4_, D_, 1.0f, 1, 1);
      lstm_point<<<dim3(16), dim3(256), 0, stream>>>(g, h + (size_t)l * B_ * D_,
                                                     c + (size_t)l * B_ * D_);
      inp = h + (size_t)l * B_ * D_;
    }
  }

  // -------- LSTM generator -> mem (32 steps) --------
  for (int t = 0; t < S_; ++t) {
    const float* inp = (t == 0) ? (const float*)xstep : (mem + (size_t)(t - 1) * B_ * D_);
    for (int l = 0; l < L_; ++l) {
      gemm(stream, inp, wih_b + (size_t)l * G4_ * D_, bih + l * G4_, bhh + l * G4_,
           nullptr, nullptr, g, B_, G4_, D_, 1.0f, 0, 1);
      gemm(stream, h + (size_t)l * B_ * D_, whh_b + (size_t)l * G4_ * D_,
           nullptr, nullptr, nullptr, nullptr, g, B_, G4_, D_, 1.0f, 1, 1);
      lstm_point<<<dim3(16), dim3(256), 0, stream>>>(g, h + (size_t)l * B_ * D_,
                                                     c + (size_t)l * B_ * D_);
      inp = h + (size_t)l * B_ * D_;
    }
    gemm(stream, h + (size_t)1 * B_ * D_, proj_b, projb, nullptr, nullptr, nullptr,
         mem + (size_t)t * B_ * D_, B_, D_, D_, 1.0f, 0, 1);
  }

  // -------- cross-attention K/V from mem (once per layer) --------
  for (int l = 0; l < L_; ++l) {
    gemm(stream, mem, caw_b + (size_t)l * QKV_ * D_ + (size_t)D_ * D_,
         cab + l * QKV_ + D_, nullptr, nullptr, nullptr,
         cakv + (size_t)l * SB_ * 2 * D_, SB_, 2 * D_, D_, 1.0f, 0, 1);
  }

  // nxt = tgt_t[0]
  copy_rows<<<dim3((B_ * D_ + 255) / 256), dim3(256), 0, stream>>>(
      tgt, S_ * D_, nxt, D_, B_, D_);

  const float scale = 22.62741699796952f;  // sqrt(512)

  // -------- autoregressive transformer decoder (32 steps) --------
  for (int i = 0; i < S_; ++i) {
    // dec_in[i] = nxt
    copy_rows<<<dim3((B_ * D_ + 255) / 256), dim3(256), 0, stream>>>(
        nxt, D_, dec + (size_t)i * B_ * D_, D_, B_, D_);
    // x = (dec @ enc_w.T + enc_b)*scale + pe
    gemm(stream, dec, enc_b, encb, nullptr, pe, nullptr, x, SB_, D_, D_, scale, 0, B_);

    for (int l = 0; l < L_; ++l) {
      // ---- self attention ----
      gemm(stream, x, saw_b + (size_t)l * QKV_ * D_, sab + l * QKV_, nullptr,
           nullptr, nullptr, qkv, SB_, QKV_, D_, 1.0f, 0, 1);
      attn_scores<<<dim3(256), dim3(256), 0, stream>>>(qkv, QKV_, 0, qkv, QKV_, D_, sc, 1);
      softmax32<<<dim3(B_ * H_ * S_), dim3(32), 0, stream>>>(sc);
      attn_apply<<<dim3(512), dim3(256), 0, stream>>>(sc, qkv, QKV_, 2 * D_, attno);
      gemm(stream, attno, saow_b + (size_t)l * D_ * D_, saob + l * D_, nullptr,
           nullptr, x, x, SB_, D_, D_, 1.0f, 0, 1);
      layernorm512<<<dim3(SB_), dim3(32), 0, stream>>>(x, n1g + l * D_, n1b + l * D_, 0);

      // ---- cross attention ----
      gemm(stream, x, caw_b + (size_t)l * QKV_ * D_, cab + l * QKV_, nullptr,
           nullptr, nullptr, qkv, SB_, D_, D_, 1.0f, 0, 1);   // q only -> qkv as (SB_,512)
      const float* kvl = cakv + (size_t)l * SB_ * 2 * D_;
      attn_scores<<<dim3(256), dim3(256), 0, stream>>>(qkv, D_, 0, kvl, 2 * D_, 0, sc, 0);
      softmax32<<<dim3(B_ * H_ * S_), dim3(32), 0, stream>>>(sc);
      attn_apply<<<dim3(512), dim3(256), 0, stream>>>(sc, kvl, 2 * D_, D_, attno);
      gemm(stream, attno, caow_b + (size_t)l * D_ * D_, caob + l * D_, nullptr,
           nullptr, x, x, SB_, D_, D_, 1.0f, 0, 1);
      layernorm512<<<dim3(SB_), dim3(32), 0, stream>>>(x, n2g + l * D_, n2b + l * D_, 0);

      // ---- feed forward ----
      gemm(stream, x, w1_b + (size_t)l * D_ * D_, b1p + l * D_, nullptr,
           nullptr, nullptr, ff, SB_, D_, D_, 1.0f, 2 /*relu*/, 1);
      gemm(stream, ff, w2_b + (size_t)l * D_ * D_, b2p + l * D_, nullptr,
           nullptr, x, x, SB_, D_, D_, 1.0f, 0, 1);
      layernorm512<<<dim3(SB_), dim3(32), 0, stream>>>(x, n3g + l * D_, n3b + l * D_, 0);
    }

    // final LN + output projection, only at position i (8 tokens)
    layernorm512<<<dim3(B_), dim3(32), 0, stream>>>(x, nfg, nfb, i * B_);
    gemm(stream, x + (size_t)i * B_ * D_, out_b, outb, nullptr, nullptr, nullptr,
         nxt, B_, D_, D_, 1.0f, 0, 1);
  }

  // output: swapaxes(dec_in, 0, 1) -> (B, S, D)
  out_transpose<<<dim3((B_ * S_ * D_ + 255) / 256), dim3(256), 0, stream>>>(dec, outp);
}
